// FusedLayerNormGRU_6674379178372
// MI455X (gfx1250) — matrix operand order repaired and checked by reading, not compiled, not run on hardware
//
#include <hip/hip_runtime.h>

// ---------------------------------------------------------------------------
// FusedLayerNormGRU for MI455X (gfx1250, wave32, WMMA bf16 16x16x32, TDM)
//   B=64, T=512, D=512, H=1024, L=3, 3H=3072
//
// Structure:
//   prep : pack weights -> bf16 WMMA-B layout; x -> bf16; one big GEMM
//          computes xg0_all = x@Wi0^T and xres_all = x@Wres^T for ALL t
//          (non-recurrent work off the sequential critical path).
//   loop : per timestep, only the recurrent GEMMs (h@Wh^T, and for layers
//          1/2 the same-step lower-layer projection) + LayerNorm-gate cell.
//          A-operands are TDM'd into LDS; weights stream from L2 (bf16,
//          ~36 MB total, resident in the 192 MB L2).
// ---------------------------------------------------------------------------

#define BB   64
#define TT   512
#define DD   512
#define HH   1024
#define LL   3
#define G3   3072           // 3*H
#define BH   (BB * HH)      // 65536
#define LBH  (LL * BB * HH) // 196608
#define EPSF 1e-5f

typedef __bf16 bf16;
typedef __attribute__((ext_vector_type(16))) __bf16 v16bf;
typedef __attribute__((ext_vector_type(8)))  __bf16 v8bf;
typedef __attribute__((ext_vector_type(8)))  float  v8f;
typedef __attribute__((ext_vector_type(4)))  unsigned int v4u;
typedef __attribute__((ext_vector_type(8)))  unsigned int v8u;
typedef __attribute__((ext_vector_type(4)))  int v4i;
typedef __attribute__((ext_vector_type(8)))  int v8i;

#if defined(__HIP_DEVICE_COMPILE__) && __has_builtin(__builtin_amdgcn_tensor_load_to_lds)
#define HAVE_TDM 1
#else
#define HAVE_TDM 0
#endif

// ---------------------------------------------------------------------------
// Weight pre-pack: fp32 row-major W[N][K] -> bf16 WMMA B-fragment layout:
//   dst[((ntile*(K/32) + ktile)*512) + lane*16 + j]
//   lane 0..15  : B[k0+j,    n0+lane]    == W[n0+lane,    k0+j]
//   lane 16..31 : B[k0+16+j, n0+lane-16] == W[n0+lane-16, k0+16+j]
// ---------------------------------------------------------------------------
__global__ void gru_pack_weights(const float* __restrict__ W, bf16* __restrict__ dst,
                                 int N, int K) {
  size_t total = (size_t)N * K;
  int ktiles = K >> 5;
  for (size_t idx = (size_t)blockIdx.x * blockDim.x + threadIdx.x; idx < total;
       idx += (size_t)gridDim.x * blockDim.x) {
    size_t tile = idx >> 9;
    int lj   = (int)(idx & 511);
    int lane = lj >> 4;
    int j    = lj & 15;
    size_t ntile = tile / ktiles;
    int    kt    = (int)(tile % ktiles);
    int g = (int)ntile * 16 + (lane & 15);
    int k = kt * 32 + (lane >> 4) * 16 + j;
    dst[idx] = (bf16)W[(size_t)g * K + k];
  }
}

__global__ void gru_cvt_bf16(const float* __restrict__ src, bf16* __restrict__ dst,
                             size_t n) {
  size_t i = (size_t)blockIdx.x * blockDim.x + threadIdx.x;
  if (i < n) dst[i] = (bf16)src[i];
}

__global__ void gru_init_h(const float* __restrict__ h0, float* __restrict__ hf,
                           bf16* __restrict__ hb, int n) {
  int i = blockIdx.x * blockDim.x + threadIdx.x;
  if (i < n) { float v = h0[i]; hf[i] = v; hb[i] = (bf16)v; }
}

__global__ void gru_copy_out(const float* __restrict__ src, float* __restrict__ dst, int n) {
  int i = blockIdx.x * blockDim.x + threadIdx.x;
  if (i < n) dst[i] = src[i];
}

// ---------------------------------------------------------------------------
// TDM issue: async-DMA a contiguous bf16 [64][K] tile into LDS.
// Descriptor per CDNA5 D# (2D tile, data_size=2B). Wave 0 issues; EXEC ignored.
// ---------------------------------------------------------------------------
__device__ __forceinline__ void tdm_issue(bf16* lds, const bf16* __restrict__ src, int K) {
#if HAVE_TDM
  if (threadIdx.x < 32) {
    unsigned ldsAddr = (unsigned)(size_t)lds;          // low 32b of LDS-aperture ptr = LDS offset
    unsigned long long ga = (unsigned long long)(size_t)src;
    v4u g0;
    g0[0] = 1u;                                        // count=1, user descriptor
    g0[1] = ldsAddr;                                   // lds_addr
    g0[2] = (unsigned)(ga & 0xffffffffull);            // global_addr[31:0]
    g0[3] = (unsigned)((ga >> 32) & 0x1ffffffull) | (2u << 30); // addr[56:32] | type=2
    v8u g1;
    g1[0] = (1u << 16);                                // data_size=1 (2 bytes), mask=0
    g1[1] = ((unsigned)K) << 16;                       // tensor_dim0[15:0]
    g1[2] = (64u << 16);                               // tensor_dim1=64
    g1[3] = ((unsigned)K) << 16;                       // tile_dim0=K
    g1[4] = 64u;                                       // tile_dim1=64, tile_dim2=0
    g1[5] = (unsigned)K;                               // tensor_dim0_stride
    g1[6] = 0u; g1[7] = 0u;
    v4i z4 = {0, 0, 0, 0};
#if __clang_major__ >= 23
    v8i z8 = {0, 0, 0, 0, 0, 0, 0, 0};
    __builtin_amdgcn_tensor_load_to_lds(g0, __builtin_bit_cast(v8i, g1), z4, z4, z8, 0);
#else
    __builtin_amdgcn_tensor_load_to_lds(g0, __builtin_bit_cast(v8i, g1), z4, z4, 0);
#endif
  }
#else
  for (int idx = threadIdx.x; idx < 64 * K; idx += (int)blockDim.x) lds[idx] = src[idx];
#endif
}

__device__ __forceinline__ void tdm_wait_all() {
#if HAVE_TDM
  if (threadIdx.x < 32) __builtin_amdgcn_s_wait_tensorcnt(0);
#endif
}

__device__ __forceinline__ v16bf cat16(v8bf lo, v8bf hi) {
  v16bf r;
#pragma unroll
  for (int i = 0; i < 8; ++i) { r[i] = lo[i]; r[i + 8] = hi[i]; }
  return r;
}

__device__ __forceinline__ v8f zero8() {
  v8f a;
#pragma unroll
  for (int i = 0; i < 8; ++i) a[i] = 0.0f;
  return a;
}

// ---------------------------------------------------------------------------
// Two 16x16 output tiles (n0, n0+16) accumulated over K: one A-fragment pair
// of ds_load_b128 feeds TWO v_wmma_f32_16x16x32_bf16 (register-level A reuse).
// ---------------------------------------------------------------------------
__device__ __forceinline__ void wmma_accum2(const bf16* lds, int K,
                                            const bf16* __restrict__ Bpack,
                                            int n0, int m0, v8f& acc0, v8f& acc1) {
  const int lane = (int)threadIdx.x & 31;
  const int half = lane >> 4;
  const int lrow = lane & 15;
  const int arow = m0 + lrow;
  const int ktiles = K >> 5;
  const bf16* b0 = Bpack + ((size_t)(n0 >> 4) * ktiles) * 512 + lane * 16;
  const bf16* b1 = Bpack + ((size_t)((n0 >> 4) + 1) * ktiles) * 512 + lane * 16;
  for (int kt = 0; kt < ktiles; ++kt) {
    const int k0 = kt << 5;
    v8bf alo = *(const v8bf*)(lds + (size_t)arow * K + k0 + half * 16);
    v8bf ahi = *(const v8bf*)(lds + (size_t)arow * K + k0 + half * 16 + 8);
    v16bf A = cat16(alo, ahi);
    v8bf b0lo = *(const v8bf*)(b0 + (size_t)kt * 512);
    v8bf b0hi = *(const v8bf*)(b0 + (size_t)kt * 512 + 8);
    v8bf b1lo = *(const v8bf*)(b1 + (size_t)kt * 512);
    v8bf b1hi = *(const v8bf*)(b1 + (size_t)kt * 512 + 8);
    acc0 = __builtin_amdgcn_wmma_f32_16x16x32_bf16(false, A, false, cat16(b0lo, b0hi),
                                                   (short)0, acc0, false, false);
    acc1 = __builtin_amdgcn_wmma_f32_16x16x32_bf16(false, A, false, cat16(b1lo, b1hi),
                                                   (short)0, acc1, false, false);
  }
}

__device__ __forceinline__ void store_tile(float* __restrict__ out, int Nld,
                                           int m0, int n0, v8f acc) {
  const int lane = (int)threadIdx.x & 31;
  const int half = lane >> 4;
  const int lrow = lane & 15;
  float* p = out + (size_t)(m0 + half * 8) * Nld + n0 + lrow;
#pragma unroll
  for (int r = 0; r < 8; ++r) p[(size_t)r * Nld] = acc[r];
}

// ---------------------------------------------------------------------------
// Recurrent GEMM: out[64][3072] = A[64][K] @ Wpack^T for up to two sources.
//   block = 512 threads = 16 waves; wave -> (mtile = w&3, 2 ntiles at w>>2);
//   grid.x = 3072/128 = 24. Both A slabs TDM'd up front, single tensorcnt wait.
// ---------------------------------------------------------------------------
__global__ __launch_bounds__(512) void gru_gemm_rec(
    const bf16* __restrict__ xB, int Kx, const bf16* __restrict__ BxPack,
    float* __restrict__ outX,
    const bf16* __restrict__ hB, int Kh, const bf16* __restrict__ BhPack,
    float* __restrict__ outH) {
  extern __shared__ bf16 smem[];
  const int wave = (int)threadIdx.x >> 5;
  const int m0 = (wave & 3) << 4;
  const int n0 = ((int)blockIdx.x * 128) + ((wave >> 2) << 5);
  bf16* xLds = smem;
  bf16* hLds = smem + (size_t)64 * (Kx > 0 ? Kx : 0);

  if (Kx > 0) tdm_issue(xLds, xB, Kx);
  tdm_issue(hLds, hB, Kh);
  tdm_wait_all();
  __syncthreads();

  if (Kx > 0) {
    v8f a0 = zero8(), a1 = zero8();
    wmma_accum2(xLds, Kx, BxPack, n0, m0, a0, a1);
    store_tile(outX, G3, m0, n0, a0);
    store_tile(outX, G3, m0, n0 + 16, a1);
  }
  {
    v8f a0 = zero8(), a1 = zero8();
    wmma_accum2(hLds, Kh, BhPack, n0, m0, a0, a1);
    store_tile(outH, G3, m0, n0, a0);
    store_tile(outH, G3, m0, n0 + 16, a1);
  }
}

// ---------------------------------------------------------------------------
// Big input-side GEMM over all timesteps (fully parallel, off critical path):
//   rows m = (b*T + t); cols [0,3072) -> xg_all (Wi0), [3072,4096) -> xres_all
//   (Wres). grid = (4096/128, 32768/64); 3072 % 128 == 0 so no slab straddles.
// ---------------------------------------------------------------------------
__global__ __launch_bounds__(512) void gru_gemm_in(
    const bf16* __restrict__ xAll, float* __restrict__ xgAll,
    float* __restrict__ xresAll,
    const bf16* __restrict__ WiP, const bf16* __restrict__ WresP) {
  extern __shared__ bf16 smem[];
  const size_t mbase = (size_t)blockIdx.y * 64;
  const int wave = (int)threadIdx.x >> 5;
  const int m0 = (wave & 3) << 4;
  const int n0 = ((int)blockIdx.x * 128) + ((wave >> 2) << 5);

  tdm_issue(smem, xAll + mbase * DD, DD);
  tdm_wait_all();
  __syncthreads();

  v8f a0 = zero8(), a1 = zero8();
  if (n0 < G3) {
    wmma_accum2(smem, DD, WiP, n0, m0, a0, a1);
    store_tile(xgAll + mbase * G3, G3, m0, n0, a0);
    store_tile(xgAll + mbase * G3, G3, m0, n0 + 16, a1);
  } else {
    const int nn = n0 - G3;
    wmma_accum2(smem, DD, WresP, nn, m0, a0, a1);
    store_tile(xresAll + mbase * HH, HH, m0, nn, a0);
    store_tile(xresAll + mbase * HH, HH, m0, nn + 16, a1);
  }
}

// ---------------------------------------------------------------------------
// Pointwise GRU cell: per batch row b, three LayerNorms + gates + residual.
//   r = sig(LN(x_r+h_r)); z = sig(LN(x_z+h_z)); n = tanh(LN(x_n + r*h_n))
//   h' = (1-z)*n + z*h + resid
// ---------------------------------------------------------------------------
__device__ __forceinline__ void block_stats(float s, float q, float* s1, float* s2,
                                            float& mean, float& inv) {
  const int tid = (int)threadIdx.x;
  __syncthreads();
  s1[tid] = s; s2[tid] = q;
  __syncthreads();
  for (int off = 128; off > 0; off >>= 1) {
    if (tid < off) { s1[tid] += s1[tid + off]; s2[tid] += s2[tid + off]; }
    __syncthreads();
  }
  mean = s1[0] * (1.0f / (float)HH);
  float var = s2[0] * (1.0f / (float)HH) - mean * mean;
  inv = rsqrtf(var + EPSF);
}

__global__ __launch_bounds__(256) void gru_cell(
    const float* __restrict__ gx, long gxStride,                 // row stride (per b)
    const float* __restrict__ gh,                                // [64][3072]
    const float* __restrict__ bi, const float* __restrict__ bh,  // [3072]
    const float* __restrict__ gr, const float* __restrict__ br,
    const float* __restrict__ gz, const float* __restrict__ bz,
    const float* __restrict__ gn, const float* __restrict__ bn,  // [1024]
    const float* __restrict__ hprev,                             // [64][1024]
    const float* __restrict__ resid, long residStride,
    float* __restrict__ hnewF, bf16* __restrict__ hnewB,
    float* __restrict__ outSeq, int t) {
  __shared__ float s1[256];
  __shared__ float s2[256];
  const int b = (int)blockIdx.x;
  const int tid = (int)threadIdx.x;
  const float* gxr = gx + (size_t)b * gxStride;
  const float* ghr = gh + (size_t)b * G3;

  float rv[4], zv[4], nv[4], a[4];
  float mean, inv;

  // r gate
  {
    float s = 0.f, q = 0.f;
#pragma unroll
    for (int j = 0; j < 4; ++j) {
      int i = tid * 4 + j;
      float v = gxr[i] + bi[i] + ghr[i] + bh[i];
      a[j] = v; s += v; q += v * v;
    }
    block_stats(s, q, s1, s2, mean, inv);
#pragma unroll
    for (int j = 0; j < 4; ++j) {
      int i = tid * 4 + j;
      float u = (a[j] - mean) * inv * gr[i] + br[i];
      rv[j] = 1.0f / (1.0f + __expf(-u));
    }
  }
  // z gate
  {
    float s = 0.f, q = 0.f;
#pragma unroll
    for (int j = 0; j < 4; ++j) {
      int i = tid * 4 + j;
      float v = gxr[HH + i] + bi[HH + i] + ghr[HH + i] + bh[HH + i];
      a[j] = v; s += v; q += v * v;
    }
    block_stats(s, q, s1, s2, mean, inv);
#pragma unroll
    for (int j = 0; j < 4; ++j) {
      int i = tid * 4 + j;
      float u = (a[j] - mean) * inv * gz[i] + bz[i];
      zv[j] = 1.0f / (1.0f + __expf(-u));
    }
  }
  // n gate (r applies elementwise to the h contribution only)
  {
    float s = 0.f, q = 0.f;
#pragma unroll
    for (int j = 0; j < 4; ++j) {
      int i = tid * 4 + j;
      float v = (gxr[2 * HH + i] + bi[2 * HH + i]) + rv[j] * (ghr[2 * HH + i] + bh[2 * HH + i]);
      a[j] = v; s += v; q += v * v;
    }
    block_stats(s, q, s1, s2, mean, inv);
#pragma unroll
    for (int j = 0; j < 4; ++j) {
      int i = tid * 4 + j;
      nv[j] = tanhf((a[j] - mean) * inv * gn[i] + bn[i]);
    }
  }
  // combine + residual, emit fp32 + bf16 (+ output sequence for top layer)
#pragma unroll
  for (int j = 0; j < 4; ++j) {
    int i = tid * 4 + j;
    size_t o = (size_t)b * HH + i;
    float h = (1.0f - zv[j]) * nv[j] + zv[j] * hprev[o] + resid[(size_t)b * residStride + i];
    hnewF[o] = h;
    hnewB[o] = (bf16)h;
    if (outSeq) outSeq[(size_t)b * TT * HH + (size_t)t * HH + i] = h;
  }
}

// ---------------------------------------------------------------------------
// Host orchestration.
// ---------------------------------------------------------------------------
extern "C" void kernel_launch(void* const* d_in, const int* in_sizes, int n_in,
                              void* d_out, int out_size, void* d_ws, size_t ws_size,
                              hipStream_t stream) {
  (void)in_sizes; (void)n_in; (void)out_size; (void)ws_size;
  const float* x    = (const float*)d_in[0];
  const float* h0   = (const float*)d_in[1];
  const float* Wi0  = (const float*)d_in[2];   // [3072,512]
  const float* Wh0  = (const float*)d_in[3];   // [3072,1024]
  const float* Wi   = (const float*)d_in[4];   // [2,3072,1024]
  const float* Wh   = (const float*)d_in[5];   // [2,3072,1024]
  const float* bi   = (const float*)d_in[6];   // [3,3072]
  const float* bh   = (const float*)d_in[7];   // [3,3072]
  const float* g_r  = (const float*)d_in[8];   // [3,1024]
  const float* b_r  = (const float*)d_in[9];
  const float* g_z  = (const float*)d_in[10];
  const float* b_z  = (const float*)d_in[11];
  const float* g_n  = (const float*)d_in[12];
  const float* b_n  = (const float*)d_in[13];
  const float* Wres = (const float*)d_in[14];  // [1024,512]

  char* ws = (char*)d_ws;
  size_t cur = 0;
  auto alloc = [&](size_t bytes) -> void* {
    void* p = ws + cur;
    cur += (bytes + 255) & ~(size_t)255;
    return p;
  };

  bf16* Wi0p  = (bf16*)alloc((size_t)G3 * DD * sizeof(bf16));
  bf16* Wh0p  = (bf16*)alloc((size_t)G3 * HH * sizeof(bf16));
  bf16* Wi1p  = (bf16*)alloc((size_t)G3 * HH * sizeof(bf16));
  bf16* Wh1p  = (bf16*)alloc((size_t)G3 * HH * sizeof(bf16));
  bf16* Wi2p  = (bf16*)alloc((size_t)G3 * HH * sizeof(bf16));
  bf16* Wh2p  = (bf16*)alloc((size_t)G3 * HH * sizeof(bf16));
  bf16* Wresp = (bf16*)alloc((size_t)HH * DD * sizeof(bf16));
  bf16* xbf   = (bf16*)alloc((size_t)BB * TT * DD * sizeof(bf16));      // 33.5 MB
  float* xgAll   = (float*)alloc((size_t)BB * TT * G3 * sizeof(float)); // 402 MB
  float* xresAll = (float*)alloc((size_t)BB * TT * HH * sizeof(float)); // 134 MB
  float* hF   = (float*)alloc((size_t)2 * LBH * sizeof(float));         // [parity][L][B][H]
  bf16*  hB   = (bf16*) alloc((size_t)2 * LBH * sizeof(bf16));
  float* gxb  = (float*)alloc((size_t)BB * G3 * sizeof(float));
  float* ghb  = (float*)alloc((size_t)BB * G3 * sizeof(float));

  // --- prep: pack weights (bf16, L2-resident), convert x, init hidden ---
  gru_pack_weights<<<1024, 256, 0, stream>>>(Wi0, Wi0p, G3, DD);
  gru_pack_weights<<<1024, 256, 0, stream>>>(Wh0, Wh0p, G3, HH);
  gru_pack_weights<<<1024, 256, 0, stream>>>(Wi, Wi1p, G3, HH);
  gru_pack_weights<<<1024, 256, 0, stream>>>(Wi + (size_t)G3 * HH, Wi2p, G3, HH);
  gru_pack_weights<<<1024, 256, 0, stream>>>(Wh, Wh1p, G3, HH);
  gru_pack_weights<<<1024, 256, 0, stream>>>(Wh + (size_t)G3 * HH, Wh2p, G3, HH);
  gru_pack_weights<<<1024, 256, 0, stream>>>(Wres, Wresp, HH, DD);
  {
    size_t n = (size_t)BB * TT * DD;
    gru_cvt_bf16<<<(unsigned)((n + 255) / 256), 256, 0, stream>>>(x, xbf, n);
  }
  gru_init_h<<<LBH / 256, 256, 0, stream>>>(h0, hF, hB, LBH);

  // --- big parallel input-side GEMM for all timesteps ---
  gru_gemm_in<<<dim3(4096 / 128, (BB * TT) / 64), 512, 64 * DD * sizeof(bf16), stream>>>(
      xbf, xgAll, xresAll, Wi0p, Wresp);

  float* outSeq = (float*)d_out;                          // [B,T,H]
  float* hFinal = (float*)d_out + (size_t)BB * TT * HH;   // [L,B,H]

  for (int t = 0; t < TT; ++t) {
    const int pi = t & 1;
    const int po = pi ^ 1;
    float* hFi = hF + (size_t)pi * LBH;
    float* hFo = hF + (size_t)po * LBH;
    bf16*  hBi = hB + (size_t)pi * LBH;
    bf16*  hBo = hB + (size_t)po * LBH;

    // ---- layer 0: only the recurrent projection h@Wh0^T is on the path ----
    gru_gemm_rec<<<G3 / 128, 512, 64 * HH * sizeof(bf16), stream>>>(
        nullptr, 0, nullptr, nullptr, hBi, HH, Wh0p, ghb);
    gru_cell<<<BB, 256, 0, stream>>>(
        xgAll + (size_t)t * G3, (long)TT * G3, ghb, bi, bh,
        g_r, b_r, g_z, b_z, g_n, b_n,
        hFi, xresAll + (size_t)t * HH, (long)TT * HH,
        hFo, hBo, nullptr, t);

    // ---- layer 1: dual-source GEMM (layer0 out + own state), resid = layer0 out ----
    gru_gemm_rec<<<G3 / 128, 512, 2 * 64 * HH * sizeof(bf16), stream>>>(
        hBo, HH, Wi1p, gxb, hBi + BH, HH, Wh1p, ghb);
    gru_cell<<<BB, 256, 0, stream>>>(
        gxb, (long)G3, ghb, bi + G3, bh + G3,
        g_r + HH, b_r + HH, g_z + HH, b_z + HH, g_n + HH, b_n + HH,
        hFi + BH, hFo, (long)HH,
        hFo + BH, hBo + BH, nullptr, t);

    // ---- layer 2: emits the output sequence ----
    gru_gemm_rec<<<G3 / 128, 512, 2 * 64 * HH * sizeof(bf16), stream>>>(
        hBo + BH, HH, Wi2p, gxb, hBi + 2 * BH, HH, Wh2p, ghb);
    gru_cell<<<BB, 256, 0, stream>>>(
        gxb, (long)G3, ghb, bi + 2 * G3, bh + 2 * G3,
        g_r + 2 * HH, b_r + 2 * HH, g_z + 2 * HH, b_z + 2 * HH, g_n + 2 * HH, b_n + 2 * HH,
        hFi + 2 * BH, hFo + BH, (long)HH,
        hFo + 2 * BH, hBo + 2 * BH, outSeq, t);
  }

  // final state: after t=511 the output parity is 0
  gru_copy_out<<<LBH / 256, 256, 0, stream>>>(hF, hFinal, LBH);
}
